// FMoELinear_10350871183910
// MI455X (gfx1250) — matrix operand order repaired and checked by reading, not compiled, not run on hardware
//
#include <hip/hip_runtime.h>

// ---------------------------------------------------------------------------
// FMoE grouped GEMM for MI455X (gfx1250, wave32, WMMA).
// out[t, o] = sum_i inp[t, i] * weight[e, o, i] + bias[e, o]
//
// Roofline: 68.7 GFLOP, ~600 MB HBM traffic -> 26 us floor @ 23.3 TB/s.
// fp32 WMMA (16x16x4) has 8x fewer FLOP/instr than bf16 WMMA (16x16x32), so
// each fp32 operand is split into bf16 hi+lo and multiplied with 3 bf16 WMMA
// products (Ah*Bh + Ah*Bl + Al*Bh), fp32 accumulate -> ~fp32 accuracy.
//
// Grid order: col-group is the FAST axis so the 4 blocks sharing an A tile
// are dispatched adjacently and the 268 MB activation matrix is read from
// HBM once (reuse via 192 MB L2), not once per col-group.
// ---------------------------------------------------------------------------

#define NUM_EXPERT   64
#define IN_FEAT      512
#define OUT_FEAT     512
#define TOTAL_TOKENS 131072

#define TILE_M 128
#define TILE_N 128
#define KC     32                         // one WMMA K step
#define NKC    (IN_FEAT / KC)             // 16
#define MAX_TILES (TOTAL_TOKENS / TILE_M + NUM_EXPERT)   // 1088
#define LDS_STRIDE 40                     // bf16 elems/row: 32 + 8 pad (80B, 20 banks)
#define LDS_TILE   (TILE_M * LDS_STRIDE)

typedef __attribute__((ext_vector_type(16))) __bf16          v16bf;
typedef __attribute__((ext_vector_type(8)))  float           v8f;
typedef __attribute__((ext_vector_type(8)))  unsigned short  ush8;
typedef __attribute__((ext_vector_type(4)))  float           f32x4;
typedef __attribute__((ext_vector_type(4)))  __bf16          bf16x4;

union Frag {
    struct { ush8 c0, c1; } p;
    v16bf v;
};

// fp32x4 -> bf16x4 hi (RNE) + bf16x4 residual. Lowers to v_cvt_pk_bf16_f32
// + v_dual_lshlrev_b32 (verified in round-2 disasm).
__device__ inline void split4(f32x4 x, bf16x4& h, bf16x4& l) {
    h = __builtin_convertvector(x, bf16x4);
    f32x4 hf = __builtin_convertvector(h, f32x4);
    l = __builtin_convertvector(x - hf, bf16x4);
}

// ---------------------------------------------------------------------------
// Scheduler: expand per-expert counts into row-tile descriptors in d_ws.
// meta[0] = tile count; tiles[i] = {expert, global_row_start, rows, 0}.
// ---------------------------------------------------------------------------
__global__ void moe_sched_kernel(const int* __restrict__ counts,
                                 int* __restrict__ meta,
                                 int4* __restrict__ tiles) {
    if (threadIdx.x != 0 || blockIdx.x != 0) return;
    int nt = 0;
    int start = 0;
    for (int e = 0; e < NUM_EXPERT; ++e) {
        int c = counts[e];
        for (int s = 0; s < c; s += TILE_M) {
            int rows = (c - s) < TILE_M ? (c - s) : TILE_M;
            tiles[nt] = make_int4(e, start + s, rows, 0);
            ++nt;
        }
        start += c;
    }
    meta[0] = nt;
}

// ---------------------------------------------------------------------------
// Tiled grouped GEMM. Block = 256 threads (8 wave32s), tile 128x128,
// K chunked by 32, double-buffered LDS (one barrier per K step).
// Waves arranged 2 (M) x 4 (N); each wave owns a 64x32 output sub-tile
// = 4x2 WMMA accumulators (v8f each).
// blockIdx.x = col-group (fast axis), blockIdx.y = row-tile index.
// ---------------------------------------------------------------------------
__global__ __launch_bounds__(256)
void moe_gemm_kernel(const float* __restrict__ inp,
                     const float* __restrict__ weight,
                     const float* __restrict__ bias,
                     float* __restrict__ out,
                     const int* __restrict__ meta,
                     const int4* __restrict__ tiles) {
    __shared__ unsigned short sAhi[2][LDS_TILE];
    __shared__ unsigned short sAlo[2][LDS_TILE];
    __shared__ unsigned short sBhi[2][LDS_TILE];
    __shared__ unsigned short sBlo[2][LDS_TILE];

    const int ntiles = meta[0];
    if ((int)blockIdx.y >= ntiles) return;

    const int4 t     = tiles[blockIdx.y];
    const int expert = t.x;
    const int row0   = t.y;           // global starting token row of this tile
    const int rows   = t.z;           // valid rows in this tile (<= 128)
    const int o0     = blockIdx.x * TILE_N;

    const int tid  = threadIdx.x;
    const int lane = tid & 31;
    const int wave = tid >> 5;
    const int wm   = wave & 1;        // M wave coord: rows [wm*64, wm*64+64)
    const int wn   = wave >> 1;       // N wave coord: cols [wn*32, wn*32+32)

    const float* __restrict__ Abase = inp + (size_t)row0 * IN_FEAT;
    const float* __restrict__ Bbase = weight + (size_t)expert * (OUT_FEAT * IN_FEAT)
                                             + (size_t)o0 * IN_FEAT;

    // Global-load assignment: 128 rows x 32 cols fp32 = 1024 float4 per matrix
    // per chunk; each of 256 threads fetches 4 float4.
    int fr[4], fc[4];
#pragma unroll
    for (int p = 0; p < 4; ++p) {
        int f = tid + p * 256;
        fr[p] = f >> 3;               // row in tile
        fc[p] = (f & 7) * 4;          // col (floats) in chunk
    }

    f32x4 ra[4], rb[4];

    auto load_chunk = [&](int k0) {
#pragma unroll
        for (int p = 0; p < 4; ++p) {
            if (fr[p] < rows) {
                ra[p] = *(const f32x4*)(Abase + (size_t)fr[p] * IN_FEAT + k0 + fc[p]);
            } else {
                ra[p] = (f32x4){};
            }
            rb[p] = *(const f32x4*)(Bbase + (size_t)fr[p] * IN_FEAT + k0 + fc[p]);
        }
    };

    auto stage = [&](int b) {
#pragma unroll
        for (int p = 0; p < 4; ++p) {
            bf16x4 ah, al, bh, bl;
            split4(ra[p], ah, al);
            split4(rb[p], bh, bl);
            const int off = fr[p] * LDS_STRIDE + fc[p];   // 8B-aligned (80B stride)
            *(bf16x4*)&sAhi[b][off] = ah;
            *(bf16x4*)&sAlo[b][off] = al;
            *(bf16x4*)&sBhi[b][off] = bh;
            *(bf16x4*)&sBlo[b][off] = bl;
        }
    };

    v8f acc[4][2];
#pragma unroll
    for (int mi = 0; mi < 4; ++mi)
#pragma unroll
        for (int ni = 0; ni < 2; ++ni)
            acc[mi][ni] = (v8f){};

    // A operand layout (ISA 7.12.2, 16-bit A 16x32): lane<16 -> row=lane,
    // K {0..7,16..23}; lane>=16 -> row=lane-16, K {8..15,24..31}.
    const int r     = lane & 15;
    const int koffA = (lane >> 4) << 3;     // 0 or 8 (bf16 units)
    // B operand layout (row striped across lanes): lane<16 -> col=lane,
    // K {0..7,16..23}; lane>=16 -> col=lane-16, K {8..15,24..31}.
    const int koffB = (lane >> 4) << 3;     // 0 or 8 (bf16 units)

    auto compute = [&](int b) {
        Frag Ah[4], Al[4], Bh[2], Bl[2];
#pragma unroll
        for (int mi = 0; mi < 4; ++mi) {
            const int base = (wm * 64 + mi * 16 + r) * LDS_STRIDE + koffA;  // 16B-aligned
            Ah[mi].p.c0 = *(const ush8*)&sAhi[b][base];
            Ah[mi].p.c1 = *(const ush8*)&sAhi[b][base + 16];
            Al[mi].p.c0 = *(const ush8*)&sAlo[b][base];
            Al[mi].p.c1 = *(const ush8*)&sAlo[b][base + 16];
        }
#pragma unroll
        for (int ni = 0; ni < 2; ++ni) {
            const int base = (wn * 32 + ni * 16 + r) * LDS_STRIDE + koffB;  // 16B-aligned
            Bh[ni].p.c0 = *(const ush8*)&sBhi[b][base];
            Bh[ni].p.c1 = *(const ush8*)&sBhi[b][base + 16];
            Bl[ni].p.c0 = *(const ush8*)&sBlo[b][base];
            Bl[ni].p.c1 = *(const ush8*)&sBlo[b][base + 16];
        }
#pragma unroll
        for (int mi = 0; mi < 4; ++mi) {
#pragma unroll
            for (int ni = 0; ni < 2; ++ni) {
                acc[mi][ni] = __builtin_amdgcn_wmma_f32_16x16x32_bf16(
                    false, Ah[mi].v, false, Bh[ni].v, (short)0, acc[mi][ni], false, false);
                acc[mi][ni] = __builtin_amdgcn_wmma_f32_16x16x32_bf16(
                    false, Ah[mi].v, false, Bl[ni].v, (short)0, acc[mi][ni], false, false);
                acc[mi][ni] = __builtin_amdgcn_wmma_f32_16x16x32_bf16(
                    false, Al[mi].v, false, Bh[ni].v, (short)0, acc[mi][ni], false, false);
            }
        }
    };

    // Software pipeline, one barrier per K step:
    //   iter kc: issue global loads for kc+1; WMMA from buf[kc&1];
    //            stage kc+1 into buf[(kc+1)&1]; barrier.
    load_chunk(0);
    stage(0);
    __syncthreads();

    for (int kc = 0; kc < NKC; ++kc) {
        if (kc + 1 < NKC) load_chunk((kc + 1) * KC);
        compute(kc & 1);
        if (kc + 1 < NKC) stage((kc + 1) & 1);
        __syncthreads();
    }

    // Store: C layout (ISA 7.12.2) — VGPR j: lanes 0-15 M=j, lanes 16-31 M=8+j;
    // N = lane&15 within the 16x16 tile.
    const float* __restrict__ bptr = bias + (size_t)expert * OUT_FEAT + o0;
    const int mlane = (lane >> 4) << 3;   // 0 or 8
    if (rows == TILE_M) {
        // Fast path: full tile, no row guards.
#pragma unroll
        for (int mi = 0; mi < 4; ++mi) {
#pragma unroll
            for (int ni = 0; ni < 2; ++ni) {
                const int n    = wn * 32 + ni * 16 + r;
                const float bv = bptr[n];
#pragma unroll
                for (int j = 0; j < 8; ++j) {
                    const int m = wm * 64 + mi * 16 + mlane + j;
                    out[(size_t)(row0 + m) * OUT_FEAT + o0 + n] = acc[mi][ni][j] + bv;
                }
            }
        }
    } else {
#pragma unroll
        for (int mi = 0; mi < 4; ++mi) {
#pragma unroll
            for (int ni = 0; ni < 2; ++ni) {
                const int n    = wn * 32 + ni * 16 + r;
                const float bv = bptr[n];
#pragma unroll
                for (int j = 0; j < 8; ++j) {
                    const int m = wm * 64 + mi * 16 + mlane + j;
                    if (m < rows) {
                        out[(size_t)(row0 + m) * OUT_FEAT + o0 + n] = acc[mi][ni][j] + bv;
                    }
                }
            }
        }
    }
}

// ---------------------------------------------------------------------------
// Inputs (setup_inputs order): inp f32[131072,512], fwd_expert_count i32[64],
// weight f32[64,512,512], bias f32[64,512]. Output f32[131072,512].
// ---------------------------------------------------------------------------
extern "C" void kernel_launch(void* const* d_in, const int* in_sizes, int n_in,
                              void* d_out, int out_size, void* d_ws, size_t ws_size,
                              hipStream_t stream) {
    const float* inp    = (const float*)d_in[0];
    const int*   counts = (const int*)  d_in[1];
    const float* weight = (const float*)d_in[2];
    const float* bias   = (const float*)d_in[3];
    float*       out    = (float*)d_out;

    int*  meta  = (int*)d_ws;                       // meta[0] = tile count
    int4* tiles = (int4*)((char*)d_ws + 16);        // up to MAX_TILES int4s

    moe_sched_kernel<<<1, 32, 0, stream>>>(counts, meta, tiles);

    // Col-group on x (fast axis) for L2 reuse of the A tile; row-tile on y.
    dim3 grid(OUT_FEAT / TILE_N, MAX_TILES);        // (4, 1088); excess tiles exit
    moe_gemm_kernel<<<grid, 256, 0, stream>>>(inp, weight, bias, out, meta, tiles);
}
